// OODGAT_38697655336976
// MI455X (gfx1250) — matrix operand order repaired and checked by reading, not compiled, not run on hardware
//
#include <hip/hip_runtime.h>

#define IN_DIM 256
#define NHEAD  4
#define HID    32
#define OUTC   8
#define NC1    (NHEAD*HID)   // 128
#define NC2    (NHEAD*OUTC)  // 32

typedef __attribute__((ext_vector_type(8)))  unsigned short v8u;
typedef __attribute__((ext_vector_type(16))) unsigned short v16u;
typedef __attribute__((ext_vector_type(16))) __bf16         v16bf;
typedef __attribute__((ext_vector_type(8)))  float          v8f;

static __device__ __forceinline__ unsigned short f2bf(float f) {
  unsigned int u = __float_as_uint(f);
  u += 0x7FFFu + ((u >> 16) & 1u);        // round-to-nearest-even
  return (unsigned short)(u >> 16);
}

// ---------------- fp32 -> bf16 convert ----------------
__global__ void k_cvt_bf16(const float* __restrict__ src,
                           unsigned short* __restrict__ dst, int n) {
  int i = blockIdx.x * blockDim.x + threadIdx.x;
  if (i < n) dst[i] = f2bf(src[i]);
}

// ---------------- pack W[K x NC] into WMMA-B tiles ----------------
// out[((ntile*(K/32)+kb)*32 + lane)*16 + j] =
//     bf16( W[(kb*32 + (lane>>4)*16 + j)*NC + ntile*16 + (lane&15)] )
__global__ void k_pack_w(const float* __restrict__ W,
                         unsigned short* __restrict__ out, int K, int NC) {
  int t = blockIdx.x * blockDim.x + threadIdx.x;
  if (t >= K * NC) return;
  int j     = t & 15;
  int lane  = (t >> 4) & 31;
  int KB    = K >> 5;
  int kb    = (t >> 9) % KB;
  int ntile = t / (KB << 9);
  int k = kb * 32 + ((lane >> 4) << 4) + j;
  int n = (ntile << 4) + (lane & 15);
  out[t] = f2bf(W[k * NC + n]);
}

// ---------------- bf16 WMMA GEMM: C[MxNC] = A[MxK] * B[KxNC] ----------------
// one 16x16 output tile per wave; K-loop of v_wmma_f32_16x16x32_bf16
__global__ void k_gemm_wmma(const unsigned short* __restrict__ Abf,
                            const unsigned short* __restrict__ Bpk,
                            float* __restrict__ C, int M, int K, int NC) {
  const int lane   = threadIdx.x & 31;
  const int wave   = threadIdx.x >> 5;
  const int nt_cnt = NC >> 4;
  const int tile   = blockIdx.x * 8 + wave;
  const int mtile  = tile / nt_cnt;
  const int ntile  = tile - mtile * nt_cnt;
  if (mtile >= (M >> 4)) return;           // wave-uniform guard (EXEC stays all-1s)

  const int row   = (mtile << 4) + (lane & 15);
  const int khalf = (lane >> 4) << 4;      // 0 or 16
  const unsigned short* arow = Abf + (size_t)row * K + khalf;
  const unsigned short* bptr = Bpk + (((size_t)ntile * (K >> 5)) << 9) + lane * 16;

  v8f acc = {};
  const int KB = K >> 5;
  for (int kb = 0; kb < KB; ++kb) {
    __builtin_prefetch(arow + kb * 32 + 64, 0, 0);
    v8u alo = *(const v8u*)(arow + kb * 32);        // K = base+0..7
    v8u ahi = *(const v8u*)(arow + kb * 32 + 8);    // K = base+8..15
    v16u av;
#pragma unroll
    for (int i = 0; i < 8; ++i) { av[i] = alo[i]; av[i + 8] = ahi[i]; }
    v16u bv = *(const v16u*)(bptr + ((size_t)kb << 9));
    union { v16u u; v16bf b; } A, B;
    A.u = av; B.u = bv;
    acc = __builtin_amdgcn_wmma_f32_16x16x32_bf16(false, A.b, false, B.b,
                                                  (short)0, acc, false, false);
  }
  const int crow0 = (mtile << 4) + ((lane >> 4) << 3);
  const int ccol  = (ntile << 4) + (lane & 15);
#pragma unroll
  for (int v = 0; v < 8; ++v)
    C[(size_t)(crow0 + v) * NC + ccol] = acc[v];
}

// ---------------- node scores: s = sigmoid(h . att) ----------------
__global__ void k_score(const float* __restrict__ h, const float* __restrict__ att,
                        float* __restrict__ s, int N, int Cc) {
  int t = blockIdx.x * blockDim.x + threadIdx.x;   // (n, head)
  if (t >= N * NHEAD) return;
  int n = t >> 2, hd = t & 3;
  const float* hp = h + (size_t)n * (NHEAD * Cc) + hd * Cc;
  const float* ap = att + hd * Cc;
  float a = 0.f;
  for (int c = 0; c < Cc; ++c) a += hp[c] * ap[c];
  s[t] = 1.f / (1.f + expf(-a));
}

// ---------------- edge weights: expw = exp(logit), z[dst] += expw ----------------
// logit in [0,1] so no max-subtraction needed for stability
__global__ void k_edge_w(const int* __restrict__ ei, const float* __restrict__ s,
                         float* __restrict__ expw, float* __restrict__ z,
                         int E, int N) {
  int t = blockIdx.x * blockDim.x + threadIdx.x;   // (e, head)
  int Etot = E + N;
  if (t >= Etot * NHEAD) return;
  int e = t >> 2, hd = t & 3;
  int src = (e < E) ? ei[e]     : (e - E);
  int dst = (e < E) ? ei[E + e] : (e - E);
  float si = s[dst * NHEAD + hd];
  float sj = s[src * NHEAD + hd];
  float w  = expf(si * sj + (1.f - si) * (1.f - sj));
  expw[t] = w;
  atomicAdd(&z[dst * NHEAD + hd], w);
}

// ---------------- message scatter: acc[dst] += h[src] * alpha ----------------
__global__ void k_edge_msg(const int* __restrict__ ei, const float* __restrict__ h,
                           const float* __restrict__ expw, const float* __restrict__ z,
                           float* __restrict__ acc, int E, int N, int NC, int Cc) {
  long t = (long)blockIdx.x * blockDim.x + threadIdx.x;  // (e, channel)
  int Etot = E + N;
  int e = (int)(t / NC);
  int c = (int)(t - (long)e * NC);
  if (e >= Etot) return;
  int src = (e < E) ? ei[e]     : (e - E);
  int dst = (e < E) ? ei[E + e] : (e - E);
  int hd  = c / Cc;
  float alpha = expw[e * NHEAD + hd] / (z[dst * NHEAD + hd] + 1e-16f);
  atomicAdd(&acc[(size_t)dst * NC + c], h[(size_t)src * NC + c] * alpha);
}

// ---------------- layer-1 epilogue: elu(acc + b1) -> bf16 ----------------
__global__ void k_fin1(const float* __restrict__ acc, const float* __restrict__ b1,
                       unsigned short* __restrict__ hmid, int N) {
  long t = (long)blockIdx.x * blockDim.x + threadIdx.x;
  if (t >= (long)N * NC1) return;
  int ch = (int)(t & (NC1 - 1));
  float v = acc[t] + b1[ch];
  v = v > 0.f ? v : (expf(v) - 1.f);
  hmid[t] = f2bf(v);
}

// ---------------- layer-2 epilogue: mean over heads + b2 ----------------
__global__ void k_fin2(const float* __restrict__ acc, const float* __restrict__ b2,
                       float* __restrict__ out, int N) {
  int t = blockIdx.x * blockDim.x + threadIdx.x;
  if (t >= N * OUTC) return;
  int n = t >> 3, c = t & 7;
  const float* a = acc + (size_t)n * NC2 + c;
  out[t] = (a[0] + a[8] + a[16] + a[24]) * 0.25f + b2[c];
}

extern "C" void kernel_launch(void* const* d_in, const int* in_sizes, int n_in,
                              void* d_out, int out_size, void* d_ws, size_t ws_size,
                              hipStream_t stream) {
  const float* x    = (const float*)d_in[0];
  const int*   ei   = (const int*)  d_in[1];
  const float* W1   = (const float*)d_in[2];
  const float* att1 = (const float*)d_in[3];
  const float* b1   = (const float*)d_in[4];
  const float* W2   = (const float*)d_in[5];
  const float* att2 = (const float*)d_in[6];
  const float* b2   = (const float*)d_in[7];
  float* out = (float*)d_out;

  const int N = in_sizes[0] / IN_DIM;   // 50000 (divisible by 16)
  const int E = in_sizes[1] / 2;        // 800000
  const int Etot = E + N;

  char* ws = (char*)d_ws;
  size_t off = 0;
  auto take = [&](size_t bytes) { size_t o = off; off = (off + bytes + 255) & ~(size_t)255; return o; };
  unsigned short* xbf  = (unsigned short*)(ws + take((size_t)N * IN_DIM * 2)); // reused as hmid_bf
  unsigned short* w1pk = (unsigned short*)(ws + take((size_t)IN_DIM * NC1 * 2));
  unsigned short* w2pk = (unsigned short*)(ws + take((size_t)NC1 * NC2 * 2));
  float* h    = (float*)(ws + take((size_t)N * NC1 * 4));   // h1, reused for h2
  float* s    = (float*)(ws + take((size_t)N * NHEAD * 4));
  float* expw = (float*)(ws + take((size_t)Etot * NHEAD * 4));
  float* z    = (float*)(ws + take((size_t)N * NHEAD * 4));
  float* acc  = (float*)(ws + take((size_t)N * NC1 * 4));   // acc1, reused for acc2
  unsigned short* hmid = xbf;                               // safe reuse

  auto cdiv = [](long a, long b) { return (int)((a + b - 1) / b); };

  // ---- layer 1 ----
  k_cvt_bf16<<<cdiv((long)N * IN_DIM, 256), 256, 0, stream>>>(x, xbf, N * IN_DIM);
  k_pack_w  <<<cdiv(IN_DIM * NC1, 256), 256, 0, stream>>>(W1, w1pk, IN_DIM, NC1);
  k_pack_w  <<<cdiv(NC1 * NC2, 256), 256, 0, stream>>>(W2, w2pk, NC1, NC2);
  k_gemm_wmma<<<cdiv((long)(N / 16) * (NC1 / 16), 8), 256, 0, stream>>>(xbf, w1pk, h, N, IN_DIM, NC1);
  k_score   <<<cdiv((long)N * NHEAD, 256), 256, 0, stream>>>(h, att1, s, N, HID);
  hipMemsetAsync(z,   0, (size_t)N * NHEAD * 4, stream);
  hipMemsetAsync(acc, 0, (size_t)N * NC1 * 4, stream);
  k_edge_w  <<<cdiv((long)Etot * NHEAD, 256), 256, 0, stream>>>(ei, s, expw, z, E, N);
  k_edge_msg<<<cdiv((long)Etot * NC1, 256), 256, 0, stream>>>(ei, h, expw, z, acc, E, N, NC1, HID);
  k_fin1    <<<cdiv((long)N * NC1, 256), 256, 0, stream>>>(acc, b1, hmid, N);

  // ---- layer 2 ----
  k_gemm_wmma<<<cdiv((long)(N / 16) * (NC2 / 16), 8), 256, 0, stream>>>(hmid, w2pk, h, N, NC1, NC2);
  k_score   <<<cdiv((long)N * NHEAD, 256), 256, 0, stream>>>(h, att2, s, N, OUTC);
  hipMemsetAsync(z,   0, (size_t)N * NHEAD * 4, stream);
  hipMemsetAsync(acc, 0, (size_t)N * NC2 * 4, stream);
  k_edge_w  <<<cdiv((long)Etot * NHEAD, 256), 256, 0, stream>>>(ei, s, expw, z, E, N);
  k_edge_msg<<<cdiv((long)Etot * NC2, 256), 256, 0, stream>>>(ei, h, expw, z, acc, E, N, NC2, OUTC);
  k_fin2    <<<cdiv((long)N * OUTC, 256), 256, 0, stream>>>(acc, b2, out, N);
}
